// Attention_64690797412790
// MI455X (gfx1250) — matrix operand order repaired and checked by reading, compiled, not run on hardware
//
#include <hip/hip_runtime.h>
#include <cmath>

typedef __attribute__((ext_vector_type(16))) _Float16 v16h;
typedef __attribute__((ext_vector_type(8)))  _Float16 v8h;
typedef __attribute__((ext_vector_type(4)))  _Float16 v4h;
typedef __attribute__((ext_vector_type(8)))  float    v8f;
typedef __attribute__((ext_vector_type(4)))  float    v4f;

#define BSZ 2
#define SEQ 2048
#define DIMD 2048
#define NHEAD 16
#define HD 128
#define HD2 64
#define MTOT (BSZ * SEQ)

static __device__ __forceinline__ v16h concat8(v8h lo, v8h hi) {
    return __builtin_shufflevector(lo, hi, 0,1,2,3,4,5,6,7,8,9,10,11,12,13,14,15);
}
static __device__ __forceinline__ v4h cvt4(v4f x) {
    return __builtin_convertvector(x, v4h);
}

// One ds_swizzle_b32 lane-xor exchange (group-of-32 mode: xor<<10 | and 0x1F)
#define SWZ_XOR(x, pat) \
    __builtin_bit_cast(float, __builtin_amdgcn_ds_swizzle(__builtin_bit_cast(int, (x)), (pat)))
#define PAT_X1 0x041F
#define PAT_X2 0x081F
#define PAT_X4 0x101F
#define PAT_X8 0x201F

// ---------------------------------------------------------------------------
// GEMM (+ optional fused RoPE epilogue): C[M,N] = A[M,K] * B[K,N]
// fp32 in/out, f16 WMMA with f32 accumulation.
// Block tile 128x128 (8 waves, 4x2 wave grid); wave tile 32x64 = 8 WMMAs/Kstep.
// As row-major [m][k]; Bs TRANSPOSED [n][k]; all fragments = 2x ds_load_b128.
// Register double-buffering: next tile's global loads overlap current WMMAs.
// ---------------------------------------------------------------------------
#define BM 128
#define BN 128
#define BK 32
#define LDA_S 40   // halfs (80 B rows, 16B-aligned)
#define LDB_T 40   // halfs

__global__ __launch_bounds__(256)
void gemm_rope_kernel(const float* __restrict__ A, const float* __restrict__ B,
                      float* __restrict__ C,
                      const float* __restrict__ fcos, const float* __restrict__ fsin,
                      int M, int N, int K, int apply_rope)
{
    __shared__ _Float16 As[BM * LDA_S];      // [m][k]
    __shared__ _Float16 Bs[BN * LDB_T];      // [n][k]  (transposed)

    const int tid  = threadIdx.x;
    const int lane = tid & 31;
    const int wave = tid >> 5;        // 0..7
    const int wm   = wave & 3;        // 32-row tile (0..3)
    const int wn   = wave >> 2;       // 64-col tile (0..1)
    const int l16  = lane & 15;
    const int hs   = (lane >> 4) & 1;
    const int kb8  = hs * 8;
    const int kb16 = hs * 16;

    const int block_m = blockIdx.y * BM;
    const int block_n = blockIdx.x * BN;

    v8f acc[2][4] = {};

    // double-buffer staging registers
    v4f  areg[4];
    float breg[4][4];

    auto load_tile = [&](int k0) {
        #pragma unroll
        for (int j = 0; j < 4; ++j) {                 // A: 128x32, 4 v4f/thread
            const int linear = j * 256 + tid;
            const int c = (linear & 7) * 4;
            const int r = linear >> 3;
            areg[j] = *(const v4f*)&A[(size_t)(block_m + r) * K + k0 + c];
        }
        #pragma unroll
        for (int j = 0; j < 4; ++j) {                 // B: 32x128, 4x4 f32/thread
            const int linear = j * 256 + tid;
            const int c = linear & 127;               // n
            const int r = (linear >> 7) * 4;          // k group
            #pragma unroll
            for (int i = 0; i < 4; ++i)
                breg[j][i] = B[(size_t)(k0 + r + i) * N + block_n + c];
        }
    };
    auto store_tile = [&]() {
        #pragma unroll
        for (int j = 0; j < 4; ++j) {
            const int linear = j * 256 + tid;
            *(v4h*)&As[(linear >> 3) * LDA_S + (linear & 7) * 4] = cvt4(areg[j]);
        }
        #pragma unroll
        for (int j = 0; j < 4; ++j) {
            const int linear = j * 256 + tid;
            const int c = linear & 127;
            const int r = (linear >> 7) * 4;
            v4h t;
            #pragma unroll
            for (int i = 0; i < 4; ++i) t[i] = (_Float16)breg[j][i];
            *(v4h*)&Bs[c * LDB_T + r] = t;
        }
    };

    load_tile(0);
    store_tile();
    __syncthreads();

    for (int k0 = 0; k0 < K; k0 += BK) {
        const bool has_next = (k0 + BK) < K;
        if (has_next) load_tile(k0 + BK);   // global loads overlap WMMAs below

        v16h af[2], bf[4];
        #pragma unroll
        for (int s = 0; s < 2; ++s) {
            const int row = wm * 32 + s * 16 + l16;
            af[s] = concat8(*(const v8h*)&As[row * LDA_S + kb8],
                            *(const v8h*)&As[row * LDA_S + 16 + kb8]);
        }
        #pragma unroll
        for (int t = 0; t < 4; ++t) {
            const int col = wn * 64 + t * 16 + l16;
            bf[t] = concat8(*(const v8h*)&Bs[col * LDB_T + kb8],
                            *(const v8h*)&Bs[col * LDB_T + 16 + kb8]);
        }
        #pragma unroll
        for (int s = 0; s < 2; ++s)
            #pragma unroll
            for (int t = 0; t < 4; ++t)
                acc[s][t] = __builtin_amdgcn_wmma_f32_16x16x32_f16(
                    false, af[s], false, bf[t], (short)0, acc[s][t], false, false);

        __syncthreads();
        if (has_next) store_tile();
        __syncthreads();
    }

    // ---- epilogue: C layout = col per lane, row per elem (+hs*8)
    #pragma unroll
    for (int s = 0; s < 2; ++s) {
        #pragma unroll
        for (int t = 0; t < 4; ++t) {
            const int col = block_n + wn * 64 + t * 16 + l16;
            #pragma unroll
            for (int r = 0; r < 8; ++r) {
                const int row = block_m + wm * 32 + s * 16 + hs * 8 + r;
                float val = acc[s][t][r];
                if (apply_rope) {
                    const int sp = row & (SEQ - 1);
                    const int d  = col & (HD - 1);
                    const int f  = d >> 1;
                    const float cv = fcos[sp * HD2 + f];
                    const float sv = fsin[sp * HD2 + f];
                    const float p  = SWZ_XOR(val, PAT_X1);   // rotary pair swap
                    val = (d & 1) ? (p * sv + val * cv) : (val * cv - p * sv);
                }
                C[(size_t)row * N + col] = val;
            }
        }
    }
}

// ---------------------------------------------------------------------------
// Flash attention: softmax(QK^T/sqrt(HD) + causal) @ V, online softmax.
// Block: 8 waves, one (b,h), 128 q rows (16/wave). K row-major, V transposed
// in LDS; all WMMA fragments are 2x ds_load_b128. Register double-buffered.
// ---------------------------------------------------------------------------
#define AT_THREADS 256
#define AT_WAVES 8
#define QB 128
#define KBLK 32
#define LDK 136   // halfs (272 B rows)
#define LDVT 40   // halfs, Vs transposed [d][k]
#define LDP 40    // halfs, per-wave P tile stride

__global__ __launch_bounds__(256)
void attn_kernel(const float* __restrict__ q, const float* __restrict__ k,
                 const float* __restrict__ v, float* __restrict__ o)
{
    __shared__ _Float16 Ks[KBLK * LDK];        // [key][d]
    __shared__ _Float16 VsT[HD * LDVT];        // [d][key]
    __shared__ _Float16 Ps[AT_WAVES * 16 * LDP];

    const int tid  = threadIdx.x;
    const int lane = tid & 31;
    const int wave = tid >> 5;
    const int l16  = lane & 15;
    const int hs   = (lane >> 4) & 1;
    const int kb8  = hs * 8;
    const int kb16 = hs * 16;

    const int b       = blockIdx.z;
    const int h       = blockIdx.y;
    const int q_block = blockIdx.x * QB;
    const int q_base  = q_block + wave * 16;

    const float scale = 0.08838834764831845f;   // 1/sqrt(128)
    const size_t hoff = (size_t)b * SEQ * DIMD + (size_t)h * HD;

    // Q as 4 A-fragments (HD = 4 K32 chunks), softmax scale folded in.
    v16h qf[4];
    {
        const float* qrow = q + hoff + (size_t)(q_base + l16) * DIMD;
        #pragma unroll
        for (int c = 0; c < 4; ++c) {
            v4f a0 = *(const v4f*)&qrow[c * 32 + kb8];
            v4f a1 = *(const v4f*)&qrow[c * 32 + kb8 + 4];
            v4f a2 = *(const v4f*)&qrow[c * 32 + 16 + kb8];
            v4f a3 = *(const v4f*)&qrow[c * 32 + 16 + kb8 + 4];
            v8h lo = __builtin_shufflevector(cvt4(a0 * scale), cvt4(a1 * scale),
                                             0,1,2,3,4,5,6,7);
            v8h hi = __builtin_shufflevector(cvt4(a2 * scale), cvt4(a3 * scale),
                                             0,1,2,3,4,5,6,7);
            qf[c] = concat8(lo, hi);
        }
    }

    v8f oacc[8] = {};
    float row_max[8], row_sum[8];
    #pragma unroll
    for (int r = 0; r < 8; ++r) { row_max[r] = -3.0e38f; row_sum[r] = 0.0f; }

    _Float16* Pw = Ps + wave * 16 * LDP;
    const int kend = q_block + QB;

    // double-buffer staging registers
    v4f  kreg[4];
    float vreg[4][4];

    auto load_kv = [&](int kb) {
        #pragma unroll
        for (int j = 0; j < 4; ++j) {              // K: 32x128, 4 v4f/thread
            const int linear = j * 256 + tid;
            const int c = (linear & 31) * 4;
            const int r = linear >> 5;
            kreg[j] = *(const v4f*)&k[hoff + (size_t)(kb + r) * DIMD + c];
        }
        #pragma unroll
        for (int j = 0; j < 4; ++j) {              // V: 32x128 transposed
            const int linear = j * 256 + tid;
            const int c = linear & 127;            // d
            const int r = (linear >> 7) * 4;       // key group
            #pragma unroll
            for (int i = 0; i < 4; ++i)
                vreg[j][i] = v[hoff + (size_t)(kb + r + i) * DIMD + c];
        }
    };
    auto store_kv = [&]() {
        #pragma unroll
        for (int j = 0; j < 4; ++j) {
            const int linear = j * 256 + tid;
            *(v4h*)&Ks[(linear >> 5) * LDK + (linear & 31) * 4] = cvt4(kreg[j]);
        }
        #pragma unroll
        for (int j = 0; j < 4; ++j) {
            const int linear = j * 256 + tid;
            const int c = linear & 127;
            const int r = (linear >> 7) * 4;
            v4h t;
            #pragma unroll
            for (int i = 0; i < 4; ++i) t[i] = (_Float16)vreg[j][i];
            *(v4h*)&VsT[c * LDVT + r] = t;
        }
    };

    load_kv(0);
    store_kv();
    __syncthreads();

    for (int kb = 0; kb < kend; kb += KBLK) {
        const bool has_next = (kb + KBLK) < kend;
        if (has_next) load_kv(kb + KBLK);          // overlaps compute below
        if (kb + 2 * KBLK < kend) {                // warm L2 two tiles ahead
            const size_t g = hoff + (size_t)(kb + 2 * KBLK + (tid >> 3)) * DIMD
                           + (size_t)(tid & 7) * 16;
            __builtin_prefetch(&k[g], 0, 0);
            __builtin_prefetch(&v[g], 0, 0);
        }

        if (kb <= q_base + 15) {   // wave-uniform causal skip
            // ---- scores: two 16-key subtiles, 4 chained WMMAs each
            v8f sc[2];
            #pragma unroll
            for (int t = 0; t < 2; ++t) {
                v8f s = {};
                const int key = t * 16 + l16;
                #pragma unroll
                for (int c = 0; c < 4; ++c) {
                    v16h bf = concat8(
                        *(const v8h*)&Ks[key * LDK + c * 32 + kb8],
                        *(const v8h*)&Ks[key * LDK + c * 32 + 16 + kb8]);
                    s = __builtin_amdgcn_wmma_f32_16x16x32_f16(
                        false, qf[c], false, bf, (short)0, s, false, false);
                }
                sc[t] = s;
            }

            // ---- causal mask + online softmax (ds_swizzle row reductions)
            float p0[8], p1[8], mloc[8];
            #pragma unroll
            for (int r = 0; r < 8; ++r) {
                const int row = q_base + hs * 8 + r;
                p0[r] = sc[0][r] + ((kb + l16)      <= row ? 0.0f : -1.0e9f);
                p1[r] = sc[1][r] + ((kb + 16 + l16) <= row ? 0.0f : -1.0e9f);
                mloc[r] = fmaxf(p0[r], p1[r]);
            }
            #pragma unroll
            for (int r = 0; r < 8; ++r) mloc[r] = fmaxf(mloc[r], SWZ_XOR(mloc[r], PAT_X1));
            #pragma unroll
            for (int r = 0; r < 8; ++r) mloc[r] = fmaxf(mloc[r], SWZ_XOR(mloc[r], PAT_X2));
            #pragma unroll
            for (int r = 0; r < 8; ++r) mloc[r] = fmaxf(mloc[r], SWZ_XOR(mloc[r], PAT_X4));
            #pragma unroll
            for (int r = 0; r < 8; ++r) mloc[r] = fmaxf(mloc[r], SWZ_XOR(mloc[r], PAT_X8));

            float resc[8];
            #pragma unroll
            for (int r = 0; r < 8; ++r) {
                const float mnew = fmaxf(row_max[r], mloc[r]);
                resc[r]    = __expf(row_max[r] - mnew);
                row_max[r] = mnew;
                p0[r] = __expf(p0[r] - mnew);
                p1[r] = __expf(p1[r] - mnew);
            }
            float lsum[8];
            #pragma unroll
            for (int r = 0; r < 8; ++r) lsum[r] = p0[r] + p1[r];
            #pragma unroll
            for (int r = 0; r < 8; ++r) lsum[r] += SWZ_XOR(lsum[r], PAT_X1);
            #pragma unroll
            for (int r = 0; r < 8; ++r) lsum[r] += SWZ_XOR(lsum[r], PAT_X2);
            #pragma unroll
            for (int r = 0; r < 8; ++r) lsum[r] += SWZ_XOR(lsum[r], PAT_X4);
            #pragma unroll
            for (int r = 0; r < 8; ++r) lsum[r] += SWZ_XOR(lsum[r], PAT_X8);
            #pragma unroll
            for (int r = 0; r < 8; ++r)
                row_sum[r] = row_sum[r] * resc[r] + lsum[r];
            #pragma unroll
            for (int nt = 0; nt < 8; ++nt)
                #pragma unroll
                for (int r = 0; r < 8; ++r)
                    oacc[nt][r] *= resc[r];

            // ---- P: C layout -> per-wave LDS -> A layout (2x ds_load_b128)
            #pragma unroll
            for (int r = 0; r < 8; ++r) {
                const int m = hs * 8 + r;
                Pw[m * LDP + l16]      = (_Float16)p0[r];
                Pw[m * LDP + 16 + l16] = (_Float16)p1[r];
            }
            v16h pa = concat8(*(const v8h*)&Pw[l16 * LDP + kb8],
                              *(const v8h*)&Pw[l16 * LDP + 16 + kb8]);

            // ---- O += P(16x32) x V(32x128): 8 WMMAs over HD subtiles
            #pragma unroll
            for (int nt = 0; nt < 8; ++nt) {
                const int d = nt * 16 + l16;
                v16h bf = concat8(*(const v8h*)&VsT[d * LDVT + kb8],
                                  *(const v8h*)&VsT[d * LDVT + 16 + kb8]);
                oacc[nt] = __builtin_amdgcn_wmma_f32_16x16x32_f16(
                    false, pa, false, bf, (short)0, oacc[nt], false, false);
            }
        }

        __syncthreads();
        if (has_next) store_kv();
        __syncthreads();
    }

    // ---- epilogue: normalize, write fp32 [b, s, h*HD + d]
    #pragma unroll
    for (int nt = 0; nt < 8; ++nt) {
        const int d = nt * 16 + l16;
        #pragma unroll
        for (int r = 0; r < 8; ++r) {
            const int row = q_base + hs * 8 + r;
            o[hoff + (size_t)row * DIMD + d] = oacc[nt][r] * (1.0f / row_sum[r]);
        }
    }
}

// ---------------------------------------------------------------------------
// Launch: 3 projection GEMMs (RoPE fused for Q,K) -> flash attention -> out GEMM
// ---------------------------------------------------------------------------
extern "C" void kernel_launch(void* const* d_in, const int* in_sizes, int n_in,
                              void* d_out, int out_size, void* d_ws, size_t ws_size,
                              hipStream_t stream)
{
    const float* x    = (const float*)d_in[0];
    const float* fcos = (const float*)d_in[1];
    const float* fsin = (const float*)d_in[2];
    // d_in[3] = dense mask (unused: causal mask computed analytically)
    const float* wq   = (const float*)d_in[4];
    const float* wk   = (const float*)d_in[5];
    const float* wv   = (const float*)d_in[6];
    const float* wo   = (const float*)d_in[7];

    const size_t plane = (size_t)MTOT * DIMD;
    float* qb = (float*)d_ws;
    float* kb = qb + plane;
    float* vb = kb + plane;
    float* ob = vb + plane;

    dim3 gg(DIMD / BN, MTOT / BM);              // 16 x 32 blocks
    gemm_rope_kernel<<<gg, 256, 0, stream>>>(x, wq, qb, fcos, fsin, MTOT, DIMD, DIMD, 1);
    gemm_rope_kernel<<<gg, 256, 0, stream>>>(x, wk, kb, fcos, fsin, MTOT, DIMD, DIMD, 1);
    gemm_rope_kernel<<<gg, 256, 0, stream>>>(x, wv, vb, fcos, fsin, MTOT, DIMD, DIMD, 0);

    dim3 ga(SEQ / QB, NHEAD, BSZ);              // 16 x 16 x 2 blocks
    attn_kernel<<<ga, 256, 0, stream>>>(qb, kb, vb, ob);

    gemm_rope_kernel<<<gg, 256, 0, stream>>>(ob, wo, (float*)d_out, fcos, fsin, MTOT, DIMD, DIMD, 0);
}
